// leAdmm_U_4707284156813
// MI455X (gfx1250) — matrix operand order repaired and checked by reading, not compiled, not run on hardware
//
#include <hip/hip_runtime.h>
#include <hip/hip_bf16.h>

// ---------------------------------------------------------------------------
// Le-ADMM-U for DiffuserCam on MI455X (gfx1250, wave32).
// Convs -> implicit-GEMM v_wmma_f32_16x16x32_f16 with LDS-staged halos and
// weight matrices pre-packed in per-lane WMMA fragment order (B fragment is a
// single 32B contiguous load per lane). FFTs -> LDS mixed-radix Stockham.
// ---------------------------------------------------------------------------

typedef __attribute__((ext_vector_type(16))) _Float16 v16h;
typedef __attribute__((ext_vector_type(8)))  float    v8f;

#define FULL_H 540
#define FULL_W 960
#define SENS_H 270
#define SENS_W 480
#define NPIX   (FULL_H * FULL_W)      // 518400
#define BATCH  8
#define BNPIX  (BATCH * NPIX)         // 4147200
#define MU1f 1e-6f
#define MU2f 1e-5f
#define MU3f 4e-5f

// ------------------------------ helpers ------------------------------------

__device__ __forceinline__ float2 cmulf(float2 a, float2 b) {
    return make_float2(a.x * b.x - a.y * b.y, a.x * b.y + a.y * b.x);
}

// 16-bit A-matrix 16x32 fragment: element e of lane L -> K index
__device__ __forceinline__ int fragA_k(int e, int lane) {
    int v = e >> 1;
    int base = (v < 4) ? (v * 2) : (16 + (v - 4) * 2);
    return base + (e & 1) + ((lane >= 16) ? 8 : 0);
}
// 16-bit B-matrix 32x16 fragment: element e of lane L -> K index
__device__ __forceinline__ int fragB_k(int e, int lane) {
    return ((e < 8) ? e : (e + 8)) + ((lane >= 16) ? 8 : 0);
}

// ------------------------------ setup kernels ------------------------------

__global__ void k_zero3(float* X, float* A1, float* A3) {
    long i = (long)blockIdx.x * blockDim.x + threadIdx.x;
    if (i < BNPIX) { X[i] = 0.f; A1[i] = 0.f; A3[i] = 0.f; }
}

// Pack conv weights directly in per-lane WMMA B-fragment order:
// Wp[chunk][lane][e] = W[k = chunk*32 + fragB_k(e,lane)][n = lane&15]
// so each lane's 16 halves are one contiguous 32-byte load.
__global__ void k_prep_w(const float* w1, const float* w2, const float* w3,
                         _Float16* Wp1, _Float16* Wp2, _Float16* Wp3) {
    int t = blockIdx.x * blockDim.x + threadIdx.x;
    if (t < 512) {                           // conv1: 1 chunk (K=9 -> 32)
        int lane = (t >> 4) & 31, e = t & 15;
        int k = fragB_k(e, lane), n = lane & 15;
        float v = (k < 9 && n < 8) ? w1[n * 9 + k] : 0.f;
        Wp1[t] = (_Float16)v;
    } else if (t < 512 + 1536) {             // conv2: 3 chunks (K=72 -> 96)
        int t2 = t - 512;
        int kc = t2 >> 9, lane = (t2 >> 4) & 31, e = t2 & 15;
        int k = kc * 32 + fragB_k(e, lane), n = lane & 15;
        float v = (k < 72) ? w2[n * 72 + k] : 0.f;
        Wp2[t2] = (_Float16)v;
    } else if (t < 512 + 1536 + 3584) {      // conv3: 7 chunks (K=216 -> 224)
        int t3 = t - 2048;
        int kc = t3 >> 9, lane = (t3 >> 4) & 31, e = t3 & 15;
        int k = kc * 32 + fragB_k(e, lane), n = lane & 15;
        float v = (k < 216 && n < 8) ? w3[n * 216 + k] : 0.f;
        Wp3[t3] = (_Float16)v;
    }
}

// C[pix] = ishift(pad_full(h)) as complex (single plane)
__global__ void k_build_psf(const float* h, float2* C) {
    int pix = blockIdx.x * blockDim.x + threadIdx.x;
    if (pix >= NPIX) return;
    int y = pix / FULL_W, x = pix % FULL_W;
    int yy = (y + FULL_H / 2) % FULL_H;
    int xx = (x + FULL_W / 2) % FULL_W;
    float v = 0.f;
    if (yy >= 135 && yy < 405 && xx >= 240 && xx < 720)
        v = h[(yy - 135) * SENS_W + (xx - 240)];
    C[pix] = make_float2(v, 0.f);
}

__global__ void k_hf_rdiv(const float2* C, float2* Hf, float* Rdiv) {
    int pix = blockIdx.x * blockDim.x + threadIdx.x;
    if (pix >= NPIX) return;
    float2 v = C[pix];
    Hf[pix] = v;
    Rdiv[pix] = 1.f / (MU1f * (v.x * v.x + v.y * v.y) + MU2f + MU3f);
}

// ------------------------------ FFT kernel ---------------------------------
// One workgroup per line. Mixed-radix Stockham in LDS (ping-pong).
// 960 = 4*4*4*3*5, 540 = 4*3*3*3*5. sign=-1 fwd, +1 inv; scale applied at store.
__global__ void k_fft(float2* C, int NLEN, int linesPerPlane, int lineMul,
                      int elemStride, float sign, float scale) {
    __shared__ float2 buf[2][FULL_W];
    int li = blockIdx.x;
    long base = (long)(li / linesPerPlane) * NPIX
              + (long)(li % linesPerPlane) * lineMul;
    int facs[5];
    if (NLEN == FULL_W) { facs[0]=4; facs[1]=4; facs[2]=4; facs[3]=3; facs[4]=5; }
    else                { facs[0]=4; facs[1]=3; facs[2]=3; facs[3]=3; facs[4]=5; }

    for (int i = threadIdx.x; i < NLEN; i += blockDim.x)
        buf[0][i] = C[base + (long)i * elemStride];
    __syncthreads();

    const float TWO_PI = 6.283185307179586f;
    int cur = 0, l = NLEN, m = 1;
    for (int s = 0; s < 5; s++) {
        int R = facs[s];
        l /= R;
        int nb = l * m;
        for (int idx = threadIdx.x; idx < nb; idx += blockDim.x) {
            int j = idx / m, k = idx - j * m;
            float2 c[5];
            for (int t = 0; t < R; t++) c[t] = buf[cur][k + m * (j + l * t)];
            for (int t = 0; t < R; t++) {
                float2 acc = make_float2(0.f, 0.f);
                for (int u = 0; u < R; u++) {
                    float ang = sign * TWO_PI * (float)((u * t) % R) / (float)R;
                    float sn, cs; __sincosf(ang, &sn, &cs);
                    acc.x += c[u].x * cs - c[u].y * sn;
                    acc.y += c[u].x * sn + c[u].y * cs;
                }
                float ang2 = sign * TWO_PI * (float)(j * t) / (float)(R * l);
                float sn2, cs2; __sincosf(ang2, &sn2, &cs2);
                buf[cur ^ 1][k + m * (R * j + t)] = cmulf(acc, make_float2(cs2, sn2));
            }
        }
        __syncthreads();
        cur ^= 1;
        m *= R;
    }
    for (int i = threadIdx.x; i < NLEN; i += blockDim.x) {
        float2 v = buf[cur][i];
        v.x *= scale; v.y *= scale;
        C[base + (long)i * elemStride] = v;
    }
}

// ------------------------------ pointwise ----------------------------------

__global__ void k_r2c(float2* C, const float* src) {
    long i = (long)blockIdx.x * blockDim.x + threadIdx.x;
    if (i < BNPIX) C[i] = make_float2(src[i], 0.f);
}
__global__ void k_realC(float* dst, const float2* C) {
    long i = (long)blockIdx.x * blockDim.x + threadIdx.x;
    if (i < BNPIX) dst[i] = C[i].x;
}
__global__ void k_cmulH(float2* C, const float2* Hf, float csign) {
    long i = (long)blockIdx.x * blockDim.x + threadIdx.x;
    if (i >= BNPIX) return;
    float2 hv = Hf[i % NPIX];
    hv.y *= csign;
    C[i] = cmulf(C[i], hv);
}
__global__ void k_cmulR(float2* C, const float* Rdiv) {
    long i = (long)blockIdx.x * blockDim.x + threadIdx.x;
    if (i >= BNPIX) return;
    float r = Rdiv[i % NPIX];
    float2 v = C[i];
    C[i] = make_float2(v.x * r, v.y * r);
}
// V = V_div * (a1 + MU1*Hop(X) + CTy)
__global__ void k_V(float* V, const float* A1, const float* HX, const float* xin) {
    long i = (long)blockIdx.x * blockDim.x + threadIdx.x;
    if (i >= BNPIX) return;
    int pix = (int)(i % NPIX); int b = (int)(i / NPIX);
    int y = pix / FULL_W, x = pix % FULL_W;
    bool in = (y >= 135 && y < 405 && x >= 240 && x < 720);
    float cty  = in ? xin[(b * SENS_H + (y - 135)) * SENS_W + (x - 240)] : 0.f;
    float vdiv = in ? (1.f / (1.f + MU1f)) : (1.f / MU1f);
    V[i] = vdiv * (A1[i] + MU1f * HX[i] + cty);
}
__global__ void k_Wstep(float* W, const float* A3, const float* X) {
    long i = (long)blockIdx.x * blockDim.x + threadIdx.x;
    if (i < BNPIX) W[i] = fmaxf(A3[i] / MU3f + X[i], 0.f);
}
__global__ void k_t2C(float2* C, const float* V, const float* A1) {
    long i = (long)blockIdx.x * blockDim.x + threadIdx.x;
    if (i < BNPIX) C[i] = make_float2(MU1f * V[i] - A1[i], 0.f);
}
// in-place: C = r = MU3*W - a3 + MU2*U + real(C)   (C currently holds HTop(t))
__global__ void k_rC(float2* C, const float* W, const float* A3, const float* U) {
    long i = (long)blockIdx.x * blockDim.x + threadIdx.x;
    if (i < BNPIX)
        C[i] = make_float2(MU3f * W[i] - A3[i] + MU2f * U[i] + C[i].x, 0.f);
}
__global__ void k_a1(float* A1, const float* HX, const float* V) {
    long i = (long)blockIdx.x * blockDim.x + threadIdx.x;
    if (i < BNPIX) A1[i] += MU1f * (HX[i] - V[i]);
}
__global__ void k_a3(float* A3, const float* X, const float* W) {
    long i = (long)blockIdx.x * blockDim.x + threadIdx.x;
    if (i < BNPIX) A3[i] += MU2f * (X[i] - W[i]);
}
__global__ void k_crop(float* out, const float* X) {
    long i = (long)blockIdx.x * blockDim.x + threadIdx.x;
    long tot = (long)BATCH * SENS_H * SENS_W;
    if (i >= tot) return;
    int x = (int)(i % SENS_W); int y = (int)((i / SENS_W) % SENS_H);
    int b = (int)(i / ((long)SENS_W * SENS_H));
    out[i] = X[(long)b * NPIX + (y + 135) * FULL_W + (x + 240)];
}

// ------------------------------ U-Net (WMMA) --------------------------------
// Blocks: 192 threads = 6 waves = 6 adjacent 16-wide tiles (96 px of one row).
// Input halos staged once into LDS (zero padding baked in), so fragment
// gathers are pure LDS reads. Weight fragments are one v16h load per lane.

// conv1: 1->8ch 3x3 on 540x960, relu; f16 output h1
__global__ void __launch_bounds__(192)
k_conv1(const float* X, const _Float16* Wm1, const float* b1, _Float16* H1) {
    const int SEGS = FULL_W / 96;           // 10
    __shared__ _Float16 sX[3 * 98];
    int seg = blockIdx.x % SEGS;
    int y   = (blockIdx.x / SEGS) % FULL_H;
    int b   = blockIdx.x / (SEGS * FULL_H);
    int x0  = seg * 96;
    __builtin_prefetch(Wm1, 0, 0);

    for (int t = threadIdx.x; t < 3 * 98; t += blockDim.x) {
        int c = t % 98, ry = t / 98;
        int yy = y - 1 + ry, xx = x0 - 1 + c;
        float v = 0.f;
        if (yy >= 0 && yy < FULL_H && xx >= 0 && xx < FULL_W)
            v = X[(b * FULL_H + yy) * FULL_W + xx];
        sX[t] = (_Float16)v;
    }
    __syncthreads();

    int wv = threadIdx.x >> 5, lane = threadIdx.x & 31, m = lane & 15;
    int xw0 = x0 + 16 * wv;
    v8f acc = {};
    v16h a;
#pragma unroll
    for (int e = 0; e < 16; e++) {
        int k = fragA_k(e, lane);
        float val = 0.f;
        if (k < 9) {
            int ky = k / 3, kx = k - ky * 3;
            val = (float)sX[ky * 98 + 16 * wv + m + kx];
        }
        a[e] = (_Float16)val;
    }
    v16h bf = ((const v16h*)Wm1)[lane];
    acc = __builtin_amdgcn_wmma_f32_16x16x32_f16(false, a, false, bf,
                                                 (short)0, acc, false, false);
    int n = lane & 15;
    if (n < 8) {
        float bias = b1[n];
#pragma unroll
        for (int r = 0; r < 8; r++) {
            int mm = r + ((lane >= 16) ? 8 : 0);
            float v = acc[r] + bias; v = v > 0.f ? v : 0.f;
            H1[((b * 8 + n) * FULL_H + y) * FULL_W + xw0 + mm] = (_Float16)v;
        }
    }
}

// 2x2 average pool: h1 [B,8,540,960] -> p [B,8,270,480]
__global__ void k_pool(const _Float16* H1, _Float16* P) {
    long i = (long)blockIdx.x * blockDim.x + threadIdx.x;
    long tot = (long)BATCH * 8 * SENS_H * SENS_W;
    if (i >= tot) return;
    int xo = (int)(i % SENS_W); int yo = (int)((i / SENS_W) % SENS_H);
    long bc = i / ((long)SENS_W * SENS_H);
    const _Float16* s = H1 + bc * NPIX;
    float v = (float)s[(2 * yo) * FULL_W + 2 * xo]
            + (float)s[(2 * yo) * FULL_W + 2 * xo + 1]
            + (float)s[(2 * yo + 1) * FULL_W + 2 * xo]
            + (float)s[(2 * yo + 1) * FULL_W + 2 * xo + 1];
    P[i] = (_Float16)(0.25f * v);
}

// conv2: 8->16ch 3x3 on 270x480, relu
__global__ void __launch_bounds__(192)
k_conv2(const _Float16* P, const _Float16* Wm2, const float* b2, _Float16* H2) {
    const int SEGS = SENS_W / 96;           // 5
    __shared__ _Float16 sP[8 * 3 * 98];
    int seg = blockIdx.x % SEGS;
    int y   = (blockIdx.x / SEGS) % SENS_H;
    int b   = blockIdx.x / (SEGS * SENS_H);
    int x0  = seg * 96;
    __builtin_prefetch(Wm2, 0, 0);

    for (int t = threadIdx.x; t < 8 * 3 * 98; t += blockDim.x) {
        int c = t % 98, ry = (t / 98) % 3, ch = t / (3 * 98);
        int yy = y - 1 + ry, xx = x0 - 1 + c;
        float v = 0.f;
        if (yy >= 0 && yy < SENS_H && xx >= 0 && xx < SENS_W)
            v = (float)P[((b * 8 + ch) * SENS_H + yy) * SENS_W + xx];
        sP[t] = (_Float16)v;
    }
    __syncthreads();

    int wv = threadIdx.x >> 5, lane = threadIdx.x & 31, m = lane & 15;
    int xw0 = x0 + 16 * wv;
    v8f acc = {};
    for (int kc = 0; kc < 3; kc++) {
        v16h a;
#pragma unroll
        for (int e = 0; e < 16; e++) {
            int k = kc * 32 + fragA_k(e, lane);
            float val = 0.f;
            if (k < 72) {
                int cin = k / 9; int rr = k - cin * 9;
                int ky = rr / 3, kx = rr - ky * 3;
                val = (float)sP[(cin * 3 + ky) * 98 + 16 * wv + m + kx];
            }
            a[e] = (_Float16)val;
        }
        v16h bf = ((const v16h*)Wm2)[kc * 32 + lane];
        acc = __builtin_amdgcn_wmma_f32_16x16x32_f16(false, a, false, bf,
                                                     (short)0, acc, false, false);
    }
    int n = lane & 15;
    float bias = b2[n];
#pragma unroll
    for (int r = 0; r < 8; r++) {
        int mm = r + ((lane >= 16) ? 8 : 0);
        float v = acc[r] + bias; v = v > 0.f ? v : 0.f;
        H2[((b * 16 + n) * SENS_H + y) * SENS_W + xw0 + mm] = (_Float16)v;
    }
}

// conv3: concat(up(h2)[16ch], h1[8ch]) -> 8ch 3x3 on 540x960, relu
// h2 halo: nearest-up means rows {y-1,y,y+1} map to 2 half-res rows starting
// at rh0=(y-1)>>1; cols map into 50 half-res cols starting at xh0=x0/2-1.
__global__ void __launch_bounds__(192)
k_conv3(const _Float16* H2, const _Float16* H1,
        const _Float16* Wm3, const float* b3, _Float16* H3) {
    const int SEGS = FULL_W / 96;           // 10
    __shared__ _Float16 sH1[8 * 3 * 98];    // 4704 B
    __shared__ _Float16 sH2[16 * 2 * 50];   // 3200 B
    int seg = blockIdx.x % SEGS;
    int y   = (blockIdx.x / SEGS) % FULL_H;
    int b   = blockIdx.x / (SEGS * FULL_H);
    int x0  = seg * 96;
    int rh0 = (y - 1) >> 1;                 // arithmetic shift: y=0 -> -1
    int xh0 = x0 / 2 - 1;
    __builtin_prefetch(Wm3, 0, 0);

    for (int t = threadIdx.x; t < 8 * 3 * 98; t += blockDim.x) {
        int c = t % 98, ry = (t / 98) % 3, ch = t / (3 * 98);
        int yy = y - 1 + ry, xx = x0 - 1 + c;
        float v = 0.f;
        if (yy >= 0 && yy < FULL_H && xx >= 0 && xx < FULL_W)
            v = (float)H1[((b * 8 + ch) * FULL_H + yy) * FULL_W + xx];
        sH1[t] = (_Float16)v;
    }
    for (int t = threadIdx.x; t < 16 * 2 * 50; t += blockDim.x) {
        int c = t % 50, rr = (t / 50) % 2, ch = t / 100;
        int rh = rh0 + rr, xh = xh0 + c;
        float v = 0.f;
        if (rh >= 0 && rh < SENS_H && xh >= 0 && xh < SENS_W)
            v = (float)H2[((b * 16 + ch) * SENS_H + rh) * SENS_W + xh];
        sH2[t] = (_Float16)v;
    }
    __syncthreads();

    int wv = threadIdx.x >> 5, lane = threadIdx.x & 31, m = lane & 15;
    int xw0 = x0 + 16 * wv;
    v8f acc = {};
    for (int kc = 0; kc < 7; kc++) {
        v16h a;
#pragma unroll
        for (int e = 0; e < 16; e++) {
            int k = kc * 32 + fragA_k(e, lane);
            float val = 0.f;
            if (k < 216) {
                int cin = k / 9; int rr = k - cin * 9;
                int ky = rr / 3, kx = rr - ky * 3;
                if (cin < 16) {
                    int yy  = y + ky - 1;
                    int xx  = x0 + 16 * wv + m + kx - 1;
                    int rhl = (yy >> 1) - rh0;            // 0 or 1
                    int col = (xx >> 1) - xh0;            // 0..49
                    val = (float)sH2[(cin * 2 + rhl) * 50 + col];
                } else {
                    val = (float)sH1[((cin - 16) * 3 + ky) * 98 + 16 * wv + m + kx];
                }
            }
            a[e] = (_Float16)val;
        }
        v16h bf = ((const v16h*)Wm3)[kc * 32 + lane];
        acc = __builtin_amdgcn_wmma_f32_16x16x32_f16(false, a, false, bf,
                                                     (short)0, acc, false, false);
    }
    int n = lane & 15;
    if (n < 8) {
        float bias = b3[n];
#pragma unroll
        for (int r = 0; r < 8; r++) {
            int mm = r + ((lane >= 16) ? 8 : 0);
            float v = acc[r] + bias; v = v > 0.f ? v : 0.f;
            H3[((b * 8 + n) * FULL_H + y) * FULL_W + xw0 + mm] = (_Float16)v;
        }
    }
}

// conv4: 1x1, 8->1 ch (pointwise)
__global__ void k_conv4(const _Float16* H3, const float* w4, const float* b4,
                        float* U) {
    long i = (long)blockIdx.x * blockDim.x + threadIdx.x;
    if (i >= BNPIX) return;
    int b = (int)(i / NPIX), pix = (int)(i % NPIX);
    float s = b4[0];
#pragma unroll
    for (int c = 0; c < 8; c++)
        s += w4[c] * (float)H3[((long)(b * 8 + c)) * NPIX + pix];
    U[i] = s;
}

// ------------------------------ host ---------------------------------------

extern "C" void kernel_launch(void* const* d_in, const int* in_sizes, int n_in,
                              void* d_out, int out_size, void* d_ws, size_t ws_size,
                              hipStream_t stream) {
    (void)in_sizes; (void)n_in; (void)out_size; (void)ws_size;
    const float* xin = (const float*)d_in[0];
    const float* h   = (const float*)d_in[1];
    const float* w1  = (const float*)d_in[2];
    const float* b1  = (const float*)d_in[3];
    const float* w2  = (const float*)d_in[4];
    const float* b2  = (const float*)d_in[5];
    const float* w3  = (const float*)d_in[6];
    const float* b3  = (const float*)d_in[7];
    const float* w4  = (const float*)d_in[8];
    const float* b4  = (const float*)d_in[9];
    float* out = (float*)d_out;

    char* base = (char*)d_ws;
    size_t off = 0;
    auto alloc = [&](size_t bytes) -> char* {
        char* p = base + off;
        off = (off + bytes + 255) & ~(size_t)255;
        return p;
    };
    float*  X    = (float*)  alloc((size_t)BNPIX * 4);
    float*  A1   = (float*)  alloc((size_t)BNPIX * 4);
    float*  A3   = (float*)  alloc((size_t)BNPIX * 4);
    float*  V    = (float*)  alloc((size_t)BNPIX * 4);
    float*  Wb   = (float*)  alloc((size_t)BNPIX * 4);
    float*  U    = (float*)  alloc((size_t)BNPIX * 4);
    float*  HX   = (float*)  alloc((size_t)BNPIX * 4);
    float2* C    = (float2*) alloc((size_t)BNPIX * 8);
    float2* Hf   = (float2*) alloc((size_t)NPIX * 8);
    float*  Rdiv = (float*)  alloc((size_t)NPIX * 4);
    _Float16* H1 = (_Float16*)alloc((size_t)BATCH * 8  * NPIX * 2);
    _Float16* P  = (_Float16*)alloc((size_t)BATCH * 8  * (NPIX / 4) * 2);
    _Float16* H2 = (_Float16*)alloc((size_t)BATCH * 16 * (NPIX / 4) * 2);
    _Float16* H3 = (_Float16*)alloc((size_t)BATCH * 8  * NPIX * 2);
    _Float16* Wm1 = (_Float16*)alloc(32  * 16 * 2);
    _Float16* Wm2 = (_Float16*)alloc(96  * 16 * 2);
    _Float16* Wm3 = (_Float16*)alloc(224 * 16 * 2);

    const int TB = 256;
    const int gBN   = BNPIX / TB;                     // 16200
    const int gPIX  = NPIX / TB;                      // 2025
    const int gC1   = BATCH * FULL_H * (FULL_W / 96); // 43200 blocks, 192 thr
    const int gC2   = BATCH * SENS_H * (SENS_W / 96); // 10800 blocks, 192 thr
    const int gPool = (BATCH * 8 * SENS_H * SENS_W) / TB;  // 32400
    const int gCrop = (BATCH * SENS_H * SENS_W) / TB;      // 4050

    auto fft2 = [&](float2* buf, int planes, int dir) {
        float sign = (dir < 0) ? -1.f : 1.f;
        float srow = (dir < 0) ? 1.f : (1.f / (float)FULL_W);
        float scol = (dir < 0) ? 1.f : (1.f / (float)FULL_H);
        // rows: len 960, 540 lines/plane, line offset 960, stride 1
        k_fft<<<dim3(planes * FULL_H), dim3(TB), 0, stream>>>(
            buf, FULL_W, FULL_H, FULL_W, 1, sign, srow);
        // cols: len 540, 960 lines/plane, line offset 1, stride 960
        k_fft<<<dim3(planes * FULL_W), dim3(TB), 0, stream>>>(
            buf, FULL_H, FULL_W, 1, FULL_W, sign, scol);
    };
    auto Hop = [&](const float* src, float* dst, float csign) {
        k_r2c<<<gBN, TB, 0, stream>>>(C, src);
        fft2(C, BATCH, -1);
        k_cmulH<<<gBN, TB, 0, stream>>>(C, Hf, csign);
        fft2(C, BATCH, +1);
        k_realC<<<gBN, TB, 0, stream>>>(dst, C);
    };

    // --- setup ---
    k_zero3<<<gBN, TB, 0, stream>>>(X, A1, A3);
    k_prep_w<<<22, TB, 0, stream>>>(w1, w2, w3, Wm1, Wm2, Wm3);
    k_build_psf<<<gPIX, TB, 0, stream>>>(h, C);
    fft2(C, 1, -1);
    k_hf_rdiv<<<gPIX, TB, 0, stream>>>(C, Hf, Rdiv);

    // --- 5 ADMM iterations ---
    for (int it = 0; it < 5; it++) {
        // U = unet(X)
        k_conv1<<<gC1, 192, 0, stream>>>(X, Wm1, b1, H1);
        k_pool <<<gPool, TB, 0, stream>>>(H1, P);
        k_conv2<<<gC2, 192, 0, stream>>>(P, Wm2, b2, H2);
        k_conv3<<<gC1, 192, 0, stream>>>(H2, H1, Wm3, b3, H3);
        k_conv4<<<gBN, TB, 0, stream>>>(H3, w4, b4, U);

        // V = V_div * (a1 + MU1*Hop(X) + CTy)
        Hop(X, HX, +1.f);
        k_V<<<gBN, TB, 0, stream>>>(V, A1, HX, xin);
        // W = max(a3/MU3 + X, 0)
        k_Wstep<<<gBN, TB, 0, stream>>>(Wb, A3, X);
        // r = MU3*W - a3 + MU2*U + HTop(MU1*V - a1)
        k_t2C<<<gBN, TB, 0, stream>>>(C, V, A1);
        fft2(C, BATCH, -1);
        k_cmulH<<<gBN, TB, 0, stream>>>(C, Hf, -1.f);   // conj(Hf)
        fft2(C, BATCH, +1);
        k_rC<<<gBN, TB, 0, stream>>>(C, Wb, A3, U);
        // X = real(ifft2(R_div * fft2(r)))
        fft2(C, BATCH, -1);
        k_cmulR<<<gBN, TB, 0, stream>>>(C, Rdiv);
        fft2(C, BATCH, +1);
        k_realC<<<gBN, TB, 0, stream>>>(X, C);
        // duals
        Hop(X, HX, +1.f);
        k_a1<<<gBN, TB, 0, stream>>>(A1, HX, V);
        k_a3<<<gBN, TB, 0, stream>>>(A3, X, Wb);
    }

    k_crop<<<gCrop, TB, 0, stream>>>(out, X);
}
